// Attention_53369263620547
// MI455X (gfx1250) — matrix operand order repaired and checked by reading, not compiled, not run on hardware
//
#include <hip/hip_runtime.h>
#include <hip/hip_bf16.h>
#include <stdint.h>

// ---------------------------------------------------------------------------
// Flash-attention-style masked self-attention for MI455X (gfx1250, wave32).
//   x: [B=8, N=2048, D=1024] fp32, mask: [B,N] bool(u8) -> out [B,N,D] fp32
// Both matmuls run on v_wmma_f32_16x16x32_bf16 (fp32 accumulate).
// ---------------------------------------------------------------------------

typedef __attribute__((ext_vector_type(16))) __bf16 v16bf;
typedef __attribute__((ext_vector_type(8)))  __bf16 v8bf;
typedef __attribute__((ext_vector_type(4)))  __bf16 v4bf;
typedef __attribute__((ext_vector_type(8)))  float  v8f;

#define B_    8
#define N_    2048
#define D_    1024
#define MQ    32            // query rows per workgroup
#define NC    32            // keys per main-loop iteration
#define QP    1032          // Qs/Ks row pitch in bf16 elems (2064 B, 16B-mult, conflict pad)
#define KTP   40            // KsT row pitch in bf16 elems (80 B, 16B-mult)
#define PP    40            // Ps row pitch in bf16 elems (80 B, 16B-mult)
#define SP    36            // Sred row pitch in f32 elems
#define EPSV  1e-7f
#define SCALE 0.03125f      // 1/sqrt(1024)

// LDS layout (bytes)
#define OFF_QS  0
#define SZ_QS   (MQ * QP * 2)          // 66048
#define OFF_KS  (OFF_QS + SZ_QS)
#define SZ_KS   (NC * QP * 2)          // 66048
#define OFF_KT  (OFF_KS + SZ_KS)
#define SZ_KT   (D_ * KTP * 2)         // 81920
#define OFF_PS  (OFF_KT + SZ_KT)
#define SZ_PS   (MQ * PP * 2)          // 2560
#define OFF_SR  (OFF_PS + SZ_PS)
#define SZ_SR   (NC * SP * 4)          // 4608
#define OFF_RS  (OFF_SR + SZ_SR)
#define OFF_MQ  (OFF_RS + MQ * 4)
#define OFF_MK  (OFF_MQ + MQ * 4)
#define SMEM_BYTES (OFF_MK + NC * 4)   // ~221.6 KB (< 320 KB/WGP)

// Assemble a v16bf WMMA operand from two contiguous 16-byte LDS chunks.
static __device__ __forceinline__ v16bf ld2(const __bf16* p0, const __bf16* p1) {
    v8bf a = *(const v8bf*)p0;
    v8bf b = *(const v8bf*)p1;
    v16bf r;
#pragma unroll
    for (int i = 0; i < 8; ++i) { r[i] = a[i]; r[i + 8] = b[i]; }
    return r;
}

__global__ __launch_bounds__(256)
void attn_fa_wmma(const float* __restrict__ x,
                  const uint8_t* __restrict__ mask,
                  float* __restrict__ out)
{
    extern __shared__ char smem[];
    __bf16* Qs  = (__bf16*)(smem + OFF_QS);   // [MQ][QP]   Q tile  (row-major over d)
    __bf16* Ks  = (__bf16*)(smem + OFF_KS);   // [NC][QP]   K tile  (row-major over d)
    __bf16* KsT = (__bf16*)(smem + OFF_KT);   // [D_][KTP]  K tile transposed (d-major)
    __bf16* Ps  = (__bf16*)(smem + OFF_PS);   // [MQ][PP]   exp(S) tile, [q][key]
    float*  Sr  = (float*) (smem + OFF_SR);   // [NC][SP]   partial S^T reduction
    float*  Rs  = (float*) (smem + OFF_RS);   // [MQ]       row sums (then 1/(s+eps))
    float*  Mq  = (float*) (smem + OFF_MQ);   // [MQ]       query mask
    float*  Mk  = (float*) (smem + OFF_MK);   // [NC]       key mask

    const int tid  = threadIdx.x;
    const int wave = tid >> 5;
    const int lane = tid & 31;
    const int ln   = lane & 15;   // low lane half index
    const int lh   = lane >> 4;   // lane half (0/1)

    const int b  = blockIdx.y;
    const int q0 = blockIdx.x * MQ;

    const float* xb = x + (size_t)b * N_ * D_;

    // ---- Load + convert Q tile (fp32 -> bf16 in LDS) ----
    {
        const float4* src = (const float4*)(xb + (size_t)q0 * D_);
        for (int i = tid; i < MQ * (D_ / 4); i += 256) {
            const int row = i >> 8;        // D_/4 == 256 float4 per row
            const int c4  = i & 255;
            float4 v = src[i];
            v4bf p;
            p[0] = (__bf16)v.x; p[1] = (__bf16)v.y;
            p[2] = (__bf16)v.z; p[3] = (__bf16)v.w;
            *(v4bf*)&Qs[row * QP + c4 * 4] = p;
        }
    }
    if (tid < MQ) {
        Mq[tid] = mask[(size_t)b * N_ + q0 + tid] ? 1.0f : 0.0f;
        Rs[tid] = 0.0f;
    }
    __syncthreads();

    // Persistent output accumulators: this wave owns d-cols [wave*128, wave*128+128)
    v8f acc[2][8];
#pragma unroll
    for (int m = 0; m < 2; ++m)
#pragma unroll
        for (int n = 0; n < 8; ++n)
#pragma unroll
            for (int i = 0; i < 8; ++i) acc[m][n][i] = 0.0f;

    // S^T work split: 8 waves = 4 tiles (kt,qt) x 2 d-halves (h)
    const int t_ = wave >> 1;
    const int kt = t_ & 1;    // key 16-row sub-tile of S^T
    const int qt = t_ >> 1;   // query 16-col sub-tile of S^T
    const int h  = wave & 1;  // d-half: [h*512, h*512+512)

    for (int j0 = 0; j0 < N_; j0 += NC) {
        // ---- Load + convert K tile; build both [key][d] and [d][key] ----
        {
            const float4* src = (const float4*)(xb + (size_t)j0 * D_);
            for (int i = tid; i < NC * (D_ / 4); i += 256) {
                const int row = i >> 8;
                const int c4  = i & 255;
                float4 v = src[i];
                __bf16 e0 = (__bf16)v.x, e1 = (__bf16)v.y;
                __bf16 e2 = (__bf16)v.z, e3 = (__bf16)v.w;
                v4bf p; p[0] = e0; p[1] = e1; p[2] = e2; p[3] = e3;
                *(v4bf*)&Ks[row * QP + c4 * 4] = p;
                const int d = c4 * 4;
                KsT[(d + 0) * KTP + row] = e0;
                KsT[(d + 1) * KTP + row] = e1;
                KsT[(d + 2) * KTP + row] = e2;
                KsT[(d + 3) * KTP + row] = e3;
            }
        }
        if (tid < NC) Mk[tid] = mask[(size_t)b * N_ + j0 + tid] ? 1.0f : 0.0f;
        if (j0 + NC < N_)  // warm L2/WGP$ for next tile (global_prefetch_b8)
            __builtin_prefetch(xb + (size_t)(j0 + NC) * D_ + tid * 16, 0, 0);
        __syncthreads();

        // ---- S^T = K * Q^T  (this wave: one 16x16 tile, half the d sweep) ----
        v8f s;
#pragma unroll
        for (int i = 0; i < 8; ++i) s[i] = 0.0f;
        {
            const __bf16* arow = &Ks[(kt * 16 + ln) * QP];   // A: K rows (M=key)
            const __bf16* brow = &Qs[(qt * 16 + ln) * QP];   // B: Q^T ([n=q][k=d])
            const int d0 = h * 512;
#pragma unroll
            for (int dd = 0; dd < 512; dd += 32) {
                const int d = d0 + dd;
                v16bf a  = ld2(&arow[d + lh * 8], &arow[d + 16 + lh * 8]);
                v16bf bb = ld2(&brow[d + lh * 8], &brow[d + 16 + lh * 8]);
                s = __builtin_amdgcn_wmma_f32_16x16x32_bf16(
                        false, a, false, bb, (short)0, s, false, false);
            }
        }
        if (h == 0) {  // publish partial for the partner wave
#pragma unroll
            for (int r = 0; r < 8; ++r)
                Sr[(kt * 16 + r + 8 * lh) * SP + (qt * 16 + ln)] = s[r];
        }
        __syncthreads();
        if (h == 1) {  // combine halves, exp+mask, row-sum, emit P tile (bf16)
            const float mqv = Mq[qt * 16 + ln];
            float lsum = 0.0f;
            v4bf plo, phi;
#pragma unroll
            for (int r = 0; r < 8; ++r) {
                const int key = kt * 16 + r + 8 * lh;
                const float sv = s[r] + Sr[key * SP + (qt * 16 + ln)];
                const float p  = __expf(sv * SCALE) * Mk[key] * mqv;
                lsum += p;
                if (r < 4) plo[r] = (__bf16)p; else phi[r - 4] = (__bf16)p;
            }
            // lane holds 8 consecutive keys for fixed q -> one contiguous 16B store
            __bf16* pdst = &Ps[(qt * 16 + ln) * PP + kt * 16 + lh * 8];
            *(v4bf*)pdst       = plo;
            *(v4bf*)(pdst + 4) = phi;
            atomicAdd(&Rs[qt * 16 + ln], lsum);   // ds_add_f32
        }
        __syncthreads();

        // ---- out += P * V   (A = Ps[q][key], B = KsT[d][key]; K-dim = 32) ----
#pragma unroll
        for (int m = 0; m < 2; ++m) {
            const __bf16* prow = &Ps[(m * 16 + ln) * PP];
            v16bf a = ld2(&prow[lh * 16], &prow[lh * 16 + 8]);
#pragma unroll
            for (int n = 0; n < 8; ++n) {
                const int dcol = wave * 128 + n * 16 + ln;
                v16bf bb = ld2(&KsT[dcol * KTP + lh * 16],
                               &KsT[dcol * KTP + lh * 16 + 8]);
                acc[m][n] = __builtin_amdgcn_wmma_f32_16x16x32_bf16(
                                false, a, false, bb, (short)0, acc[m][n],
                                false, false);
            }
        }
        __syncthreads();  // protect Ks/KsT/Ps before next tile load
    }

    // ---- Epilogue: divide by (rowsum + eps), store fp32 ----
    if (tid < MQ) Rs[tid] = 1.0f / (Rs[tid] + EPSV);
    __syncthreads();

    float* ob = out + (size_t)b * N_ * D_;
#pragma unroll
    for (int m = 0; m < 2; ++m) {
        float inv[8];
#pragma unroll
        for (int r = 0; r < 8; ++r) inv[r] = Rs[m * 16 + r + 8 * lh];
#pragma unroll
        for (int n = 0; n < 8; ++n) {
            const int dcol = wave * 128 + n * 16 + ln;
#pragma unroll
            for (int r = 0; r < 8; ++r) {
                const int q = q0 + m * 16 + r + 8 * lh;
                ob[(size_t)q * D_ + dcol] = acc[m][n][r] * inv[r];
            }
        }
    }
}

extern "C" void kernel_launch(void* const* d_in, const int* in_sizes, int n_in,
                              void* d_out, int out_size, void* d_ws, size_t ws_size,
                              hipStream_t stream) {
    (void)in_sizes; (void)n_in; (void)d_ws; (void)ws_size; (void)out_size;
    const float*   x    = (const float*)d_in[0];
    const uint8_t* mask = (const uint8_t*)d_in[1];
    float*         out  = (float*)d_out;

    static_assert(SMEM_BYTES <= 320 * 1024, "LDS budget exceeded");
    hipFuncSetAttribute((const void*)attn_fa_wmma,
                        hipFuncAttributeMaxDynamicSharedMemorySize, SMEM_BYTES);

    dim3 grid(N_ / MQ, B_);   // 64 x 8 = 512 workgroups
    dim3 block(256);          // 8 wave32
    attn_fa_wmma<<<grid, block, SMEM_BYTES, stream>>>(x, mask, out);
}
